// Head_16922171146880
// MI455X (gfx1250) — matrix operand order repaired and checked by reading, not compiled, not run on hardware
//
#include <hip/hip_runtime.h>

// Single attention head, fp32 in/out, bf16 WMMA compute (CDNA5 / gfx1250).
//   x: [4,4096,1024] f32; Wq/Wk/Wv: [1024,128] f32; out: [4,4096,128] f32
// Kernel 1 (fused): q/k/v = x @ W in ONE pass over x (x staged once per tile,
//   12 waves = 4 row-groups x 3 matrices); q/k row-major bf16, v stored
//   TRANSPOSED (vT[b][hd][row]) so kernel 2 can DMA V without transposing.
// Kernel 2: causal flash attention; K/V tiles staged via TDM
//   (tensor_load_to_lds, TENSORcnt double-buffering) -> f32 out

#define B_BATCH 4
#define T_SEQ   4096
#define C_EMBD  1024
#define HS      128
#define SCALE   0.08838834764831845f   // 1/sqrt(128)
#define NEG_BIG (-1.0e30f)

typedef __attribute__((ext_vector_type(16))) __bf16          v16bf;
typedef __attribute__((ext_vector_type(16))) unsigned short  v16u;
typedef __attribute__((ext_vector_type(8)))  unsigned short  v8u;   // 16B
typedef __attribute__((ext_vector_type(8)))  float           v8f;
typedef __attribute__((ext_vector_type(4)))  float           v4f;
typedef __attribute__((ext_vector_type(4)))  unsigned int    v4u;   // D# group0
typedef __attribute__((ext_vector_type(8)))  unsigned int    v8ui;  // D# group1

union Frag {
  v16u v;
  v8u  h8[2];
};

__device__ inline unsigned short f32_bf16(float f) {
  unsigned int u = __builtin_bit_cast(unsigned int, f);
  u += 0x7FFFu + ((u >> 16) & 1u);            // round-to-nearest-even
  return (unsigned short)(u >> 16);
}

__device__ inline v8f wmma_bf16(const Frag& a, const Frag& b, v8f c) {
  // D(16x16 f32) = A(16x32 bf16) * B(32x16 bf16) + C
  return __builtin_amdgcn_wmma_f32_16x16x32_bf16(
      false, __builtin_bit_cast(v16bf, a.v),
      false, __builtin_bit_cast(v16bf, b.v),
      (short)0, c, false, false);
}

// TDM: issue a 2-D tensor tile load into LDS.  group0 = {count=1, lds_addr,
// global_addr[31:0], global_addr[56:32] | type=2}; group1 passed by caller.
// VADDR2/VADDR3 omitted (NULL) -> 2-D descriptor.  Tracked by TENSORcnt.
__device__ inline void tdm_load_2d(unsigned lds_off, unsigned long long gaddr,
                                   v8ui g1) {
  v4u g0;
  g0[0] = 1u;                                   // count=1, no gather
  g0[1] = lds_off;                              // LDS byte address
  g0[2] = (unsigned)gaddr;                      // global_addr[31:0]
  g0[3] = (unsigned)(gaddr >> 32) | (2u << 30); // global_addr[56:32] | type=2
  asm volatile("tensor_load_to_lds %0, %1" :: "s"(g0), "s"(g1) : "memory");
}

// ---------------------------------------------------------------------------
// Kernel 1: fused q/k/v projection.  grid = 16384/64 blocks, block = 384
// (12 waves): wave w -> matrix (w>>2), row-group (w&3).  Per K-step the
// 64x32 x tile is staged ONCE (shared by all 3 matrices) plus all three
// 32x128 W tiles (transposed for contiguous B-frag gathers).
// ---------------------------------------------------------------------------
__global__ __launch_bounds__(384)
void qkv_proj_kernel(const float* __restrict__ x,
                     const float* __restrict__ Wq,
                     const float* __restrict__ Wk,
                     const float* __restrict__ Wv,
                     unsigned short* __restrict__ qkv) {
  __shared__ unsigned short Xs[64][40];        // 64 rows x 32 K, +8 pad
  __shared__ unsigned short Wt[3][128][40];    // 3 x (128 N x 32 K), +8 pad
  __shared__ unsigned short Ts[128][72];       // v-transpose: [hd][row]+pad

  const int tid  = threadIdx.x;
  const int wv   = tid >> 5;        // 0..11
  const int lane = tid & 31;
  const int nlan = lane & 15;       // M for A-frag, N for B-frag
  const int half = lane >> 4;
  const int mat  = wv >> 2;         // 0=q, 1=k, 2=v
  const int rg   = wv & 3;          // row-group within the 64-row tile

  const int row0 = blockIdx.x * 64;
  const size_t SZ = (size_t)B_BATCH * T_SEQ * HS;

  v8f acc[8];
  #pragma unroll
  for (int nt = 0; nt < 8; ++nt) acc[nt] = (v8f){0.f,0.f,0.f,0.f,0.f,0.f,0.f,0.f};

  for (int kb = 0; kb < C_EMBD; kb += 32) {
    __syncthreads();
    // Stage X tile 64x32 f32 -> bf16 once (512 float4 chunks over 384 threads)
    #pragma unroll
    for (int j = 0; j < 2; ++j) {
      int vi = tid + 384 * j;
      if (vi < 512) {
        int rr = vi >> 3;
        int c4 = vi & 7;
        v4f xv = *(const v4f*)(x + (size_t)(row0 + rr) * C_EMBD + kb + c4 * 4);
        #pragma unroll
        for (int e = 0; e < 4; ++e) Xs[rr][c4 * 4 + e] = f32_bf16(xv[e]);
      }
    }
    // Stage all three W tiles 32x128 f32 -> bf16 transposed (3072 float4 / 384)
    #pragma unroll
    for (int j = 0; j < 8; ++j) {
      int vi = tid + 384 * j;             // 0..3071
      int m  = vi >> 10;                  // matrix
      int rr = (vi >> 5) & 31;            // k row
      int c4 = vi & 31;                   // n/4
      const float* W = (m == 0) ? Wq : (m == 1) ? Wk : Wv;
      v4f wl = *(const v4f*)(W + (size_t)(kb + rr) * HS + c4 * 4);
      #pragma unroll
      for (int e = 0; e < 4; ++e) Wt[m][c4 * 4 + e][rr] = f32_bf16(wl[e]);
    }
    __syncthreads();

    // A fragment: lane=M, elem i -> k = (i>=8?16:0) + half*8 + (i&7)
    Frag a;
    a.h8[0] = *(const v8u*)&Xs[rg * 16 + nlan][half * 8];
    a.h8[1] = *(const v8u*)&Xs[rg * 16 + nlan][16 + half * 8];

    #pragma unroll
    for (int nt = 0; nt < 8; ++nt) {
      // B fragment: lane=N, elem i -> k = (i>=8?16:0) + half*8 + (i&7)
      Frag b;
      b.h8[0] = *(const v8u*)&Wt[mat][nt * 16 + nlan][half * 8];
      b.h8[1] = *(const v8u*)&Wt[mat][nt * 16 + nlan][16 + half * 8];
      acc[nt] = wmma_bf16(a, b, acc[nt]);
    }
  }

  if (mat != 2) {
    // q, k: row-major.  C/D layout: VGPR r holds M = r + 8*half, lane = N.
    unsigned short* out = qkv + (size_t)mat * SZ;
    #pragma unroll
    for (int nt = 0; nt < 8; ++nt) {
      #pragma unroll
      for (int r = 0; r < 8; ++r) {
        int row = row0 + rg * 16 + r + half * 8;
        out[(size_t)row * HS + nt * 16 + nlan] = f32_bf16(acc[nt][r]);
      }
    }
  } else {
    // v: stage transposed 64x128 tile in LDS (rows r..r+7 contiguous)
    #pragma unroll
    for (int nt = 0; nt < 8; ++nt) {
      v8u pk;
      #pragma unroll
      for (int r = 0; r < 8; ++r) pk[r] = f32_bf16(acc[nt][r]);
      *(v8u*)&Ts[nt * 16 + nlan][rg * 16 + half * 8] = pk;
    }
  }
  __syncthreads();   // all threads converge here (both branches)

  if (tid < 128) {
    // emit vT[b][hd][row]: thread t = hd row, 64 contiguous rows per thread
    const int b  = row0 >> 12;       // row0 / 4096
    const int rb = row0 & 4095;
    unsigned short* vT = qkv + 2 * SZ + (size_t)b * HS * T_SEQ
                       + (size_t)tid * T_SEQ + rb;
    #pragma unroll
    for (int j = 0; j < 8; ++j)
      *(v8u*)(vT + j * 8) = *(const v8u*)&Ts[tid][j * 8];
  }
}

// ---------------------------------------------------------------------------
// Kernel 2: causal flash attention.  grid = (4096/64, 4), block = 128.
// Each wave owns 16 query rows; K (row-major) and V (pre-transposed) 32-key
// chunks are DMA'd into double-buffered LDS by the Tensor Data Mover with
// hardware row padding; wave 0 issues + waits TENSORcnt, barrier publishes.
// ---------------------------------------------------------------------------
__global__ __launch_bounds__(128)
void flash_attn_kernel(const unsigned short* __restrict__ qg,
                       const unsigned short* __restrict__ kg,
                       const unsigned short* __restrict__ vg,   // vT[b][hd][row]
                       float* __restrict__ out) {
  __shared__ unsigned short Ks2[2][32][136];   // 32 keys x 128 hd, +8 pad (TDM pad)
  __shared__ unsigned short Vt2[2][128][40];   // 128 hd x 32 keys, +8 pad (TDM pad)
  __shared__ unsigned short Ps[4][16][32];     // per-wave P re-layout buffer

  const int tid  = threadIdx.x;
  const int wv   = tid >> 5;
  const int lane = tid & 31;
  const int nlan = lane & 15;
  const int half = lane >> 4;

  const int b  = blockIdx.y;
  const int qb = blockIdx.x * 64;
  const size_t seq0 = (size_t)b * T_SEQ;

  // K tensor descriptor group1: u64 elements (data_size=3); row = 32 u64
  // (256B = 64 DW -> pad_interval code 5), pad 16B (4 DW -> code 3) => LDS
  // row stride 272B, matching Ks2[.][key][136].  tensor 4096 rows x 32 u64.
  const v8ui KG1 = { (3u << 16) | (1u << 20) | (5u << 22) | (3u << 25),
                     32u << 16,      // tensor_dim0 = 32 u64
                     4096u << 16,    // tensor_dim1 = 4096 rows
                     32u << 16,      // tile_dim0 = 32
                     32u,            // tile_dim1 = 32, tile_dim2 = 0
                     32u,            // tensor_dim0_stride = 32 u64
                     0u, 0u };
  // V tensor (vT, per batch 128 x 4096): row = 8 u64 (64B = 16 DW -> code 3),
  // pad 4 DW (code 3) => LDS row stride 80B, matching Vt2[.][hd][40].
  const v8ui VG1 = { (3u << 16) | (1u << 20) | (3u << 22) | (3u << 25),
                     1024u << 16,    // tensor_dim0 = 1024 u64 (4096 bf16)
                     128u << 16,     // tensor_dim1 = 128 hd rows
                     8u << 16,       // tile_dim0 = 8 u64 (32 keys)
                     128u,           // tile_dim1 = 128
                     1024u,          // tensor_dim0_stride = 1024 u64
                     0u, 0u };

  // Q fragments for this wave's 16 rows (A-layout, 4 chunks of K=32 over hd)
  Frag qa[4];
  {
    const unsigned short* qrow = qg + (seq0 + qb + wv * 16 + nlan) * HS;
    #pragma unroll
    for (int kc = 0; kc < 4; ++kc) {
      qa[kc].h8[0] = *(const v8u*)(qrow + kc * 32 + half * 8);
      qa[kc].h8[1] = *(const v8u*)(qrow + kc * 32 + 16 + half * 8);
    }
  }

  v8f accO[8];
  float mrow[8], lrow[8];
  #pragma unroll
  for (int i = 0; i < 8; ++i) {
    accO[i] = (v8f){0.f,0.f,0.f,0.f,0.f,0.f,0.f,0.f};
    mrow[i] = NEG_BIG;
    lrow[i] = 0.f;
  }

  const int qmaxw = qb + wv * 16 + 15;
  const int nch   = (qb + 64) >> 5;          // 32-key chunks

  if (wv == 0) {
    // prologue: DMA chunk 0 into buffer 0
    tdm_load_2d((unsigned)(unsigned long long)&Ks2[0][0][0],
                (unsigned long long)(const void*)(kg + (seq0 + 0) * HS), KG1);
    tdm_load_2d((unsigned)(unsigned long long)&Vt2[0][0][0],
                (unsigned long long)(const void*)(vg + (size_t)b * HS * T_SEQ + 0),
                VG1);
  }

  for (int i = 0; i < nch; ++i) {
    const int kb = i << 5;
    if (wv == 0) {
      if (i + 1 < nch) {
        const int kb1 = (i + 1) << 5;
        const int bf1 = (i + 1) & 1;
        // Safe: trailing barrier of iter i-1 means chunk i-1 compute is done.
        tdm_load_2d((unsigned)(unsigned long long)&Ks2[bf1][0][0],
                    (unsigned long long)(const void*)(kg + (seq0 + kb1) * HS),
                    KG1);
        tdm_load_2d((unsigned)(unsigned long long)&Vt2[bf1][0][0],
                    (unsigned long long)(const void*)
                        (vg + (size_t)b * HS * T_SEQ + kb1),
                    VG1);
        __builtin_amdgcn_s_wait_tensorcnt(2);  // in-order: chunk i has landed
      } else {
        __builtin_amdgcn_s_wait_tensorcnt(0);
      }
    }
    __syncthreads();   // publish chunk i's LDS to all waves

    if (kb <= qmaxw) {   // wave-uniform: EXEC stays all-ones (WMMA req.)
      const unsigned short (*Ks)[136] = Ks2[i & 1];
      const unsigned short (*Vt)[40]  = Vt2[i & 1];

      // S = Q @ K^T for two 16-key subtiles
      float st[2][8];
      #pragma unroll
      for (int t = 0; t < 2; ++t) {
        v8f s = (v8f){0.f,0.f,0.f,0.f,0.f,0.f,0.f,0.f};
        #pragma unroll
        for (int kc = 0; kc < 4; ++kc) {
          Frag bk;   // B-frag: lane = key (N), elems = hd (table order)
          bk.h8[0] = *(const v8u*)&Ks[t * 16 + nlan][kc * 32 + half * 8];
          bk.h8[1] = *(const v8u*)&Ks[t * 16 + nlan][kc * 32 + 16 + half * 8];
          s = wmma_bf16(qa[kc], bk, s);
        }
        const int key = kb + t * 16 + nlan;
        #pragma unroll
        for (int r = 0; r < 8; ++r) {
          const int qrow = qb + wv * 16 + r + half * 8;
          float sv = s[r] * SCALE;
          st[t][r] = (key <= qrow) ? sv : NEG_BIG;   // causal mask
        }
      }

      // Online softmax: row reductions inside each 16-lane group
      float alpha[8];
      #pragma unroll
      for (int r = 0; r < 8; ++r) {
        float v0 = fmaxf(st[0][r], st[1][r]);
        #pragma unroll
        for (int off = 1; off < 16; off <<= 1)
          v0 = fmaxf(v0, __shfl_xor(v0, off, 32));
        const float mnew = fmaxf(mrow[r], v0);
        const float al   = __expf(mrow[r] - mnew);
        mrow[r] = mnew;
        const float p0 = __expf(st[0][r] - mnew);
        const float p1 = __expf(st[1][r] - mnew);
        float ls = p0 + p1;
        #pragma unroll
        for (int off = 1; off < 16; off <<= 1)
          ls += __shfl_xor(ls, off, 32);
        lrow[r] = lrow[r] * al + ls;
        alpha[r] = al;
        Ps[wv][r + half * 8][nlan]      = f32_bf16(p0);
        Ps[wv][r + half * 8][16 + nlan] = f32_bf16(p1);
      }

      #pragma unroll
      for (int nt = 0; nt < 8; ++nt)
        #pragma unroll
        for (int r = 0; r < 8; ++r) accO[nt][r] *= alpha[r];

      // Reload P in A-layout (same-wave LDS ops are ordered)
      Frag pa;
      pa.h8[0] = *(const v8u*)&Ps[wv][nlan][half * 8];
      pa.h8[1] = *(const v8u*)&Ps[wv][nlan][16 + half * 8];

      // O += P @ V
      #pragma unroll
      for (int nt = 0; nt < 8; ++nt) {
        Frag bv;   // B-frag: lane = hd (N), elems = key (table order)
        bv.h8[0] = *(const v8u*)&Vt[nt * 16 + nlan][half * 8];
        bv.h8[1] = *(const v8u*)&Vt[nt * 16 + nlan][16 + half * 8];
        accO[nt] = wmma_bf16(pa, bv, accO[nt]);
      }
    }
    __syncthreads();   // chunk i compute done; buffer may be overwritten
  }

  // Normalize and store f32 output
  #pragma unroll
  for (int nt = 0; nt < 8; ++nt) {
    #pragma unroll
    for (int r = 0; r < 8; ++r) {
      const int row = qb + wv * 16 + r + half * 8;
      out[(seq0 + row) * HS + nt * 16 + nlan] = accO[nt][r] / lrow[r];
    }
  }
}

// ---------------------------------------------------------------------------
extern "C" void kernel_launch(void* const* d_in, const int* in_sizes, int n_in,
                              void* d_out, int out_size, void* d_ws, size_t ws_size,
                              hipStream_t stream) {
  const float* x  = (const float*)d_in[0];
  const float* Wq = (const float*)d_in[1];
  const float* Wk = (const float*)d_in[2];
  const float* Wv = (const float*)d_in[3];

  // Workspace: q,k row-major bf16; vT transposed bf16.  3 * 16384*128*2B.
  unsigned short* qkv = (unsigned short*)d_ws;
  const size_t SZ = (size_t)B_BATCH * T_SEQ * HS;

  dim3 g1((B_BATCH * T_SEQ) / 64);
  qkv_proj_kernel<<<g1, dim3(384), 0, stream>>>(x, Wq, Wk, Wv, qkv);

  dim3 g2(T_SEQ / 64, B_BATCH);
  flash_attn_kernel<<<g2, dim3(128), 0, stream>>>(qkv, qkv + SZ, qkv + 2 * SZ,
                                                  (float*)d_out);
}
